// RNN_50182397886603
// MI455X (gfx1250) — matrix operand order repaired and checked by reading, not compiled, not run on hardware
//
#include <hip/hip_runtime.h>
#include <hip/hip_bf16.h>

typedef __attribute__((ext_vector_type(16))) _Float16 v16h;
typedef __attribute__((ext_vector_type(8)))  _Float16 v8h;
typedef __attribute__((ext_vector_type(8)))  float    v8f;
typedef __attribute__((ext_vector_type(2)))  float    v2f;

// Problem constants
// B=64, T=1024, IN=9, H=32, 4H=128, fc1: 2048 x 65536, fc2: 9216 x 2048

#define LOG2E  1.4426950408889634f
#define LOG2E2 2.8853900817779268f

__device__ __forceinline__ float fast_exp2(float x) {
#if __has_builtin(__builtin_amdgcn_exp2f)
    return __builtin_amdgcn_exp2f(x);        // raw v_exp_f32
#else
    return exp2f(x);
#endif
}

__device__ __forceinline__ float fast_rcp(float x) {
#if __has_builtin(__builtin_amdgcn_rcpf)
    return __builtin_amdgcn_rcpf(x);         // raw v_rcp_f32 (no div chain)
#else
    return 1.0f / x;
#endif
}

__device__ __forceinline__ float sigf(float x) {
    // sigmoid: saturates correctly through exp2->inf/0
    return fast_rcp(1.0f + fast_exp2(x * -LOG2E));
}

__device__ __forceinline__ float tanh_fast(float x) {
#if __has_builtin(__builtin_amdgcn_tanhf)
    return __builtin_amdgcn_tanhf(x);        // gfx1250 native v_tanh_f32
#elif __has_builtin(__builtin_amdgcn_tanh_f32)
    return __builtin_amdgcn_tanh_f32(x);
#else
    // tanh(x) = 1 - 2/(exp(2x)+1); branch-free, saturates via inf/0
    const float e = fast_exp2(x * LOG2E2);
    return 1.0f - 2.0f * fast_rcp(e + 1.0f);
#endif
}

// ---------------------------------------------------------------------------
// Kernel 1: layer-0 input projections (K=9, too skinny for WMMA).
// xp[dir][t*64+b][c] = sum_d x[b,t,d]*w_ih[c,d] + b_ih[c] + b_hh[c]
// ---------------------------------------------------------------------------
__global__ __launch_bounds__(256) void xp0_kernel(
    const float* __restrict__ x,
    const float* __restrict__ wf, const float* __restrict__ bif, const float* __restrict__ bhf,
    const float* __restrict__ wr, const float* __restrict__ bir, const float* __restrict__ bhr,
    float* __restrict__ xpf, float* __restrict__ xpr)
{
    const int idx = blockIdx.x * 256 + threadIdx.x;   // 2 * 65536 * 128 = 2^24
    const int dir = idx >> 23;
    const int rem = idx & ((1 << 23) - 1);
    const int r   = rem >> 7;          // t*64 + b
    const int c   = rem & 127;
    const int b   = r & 63, t = r >> 6;
    const float* w = dir ? wr : wf;
    float s = dir ? (bir[c] + bhr[c]) : (bif[c] + bhf[c]);
    const float* xr   = x + (b * 1024 + t) * 9;
    const float* wrow = w + c * 9;
    #pragma unroll
    for (int d = 0; d < 9; ++d) s += xr[d] * wrow[d];
    (dir ? xpr : xpf)[r * 128 + c] = s;
}

// ---------------------------------------------------------------------------
// Kernel 2: LSTM scan. One block per (dir, batch-tile-of-16). 64 threads =
// 2 waves; wave = hidden-half. Each wave issues 4x v_wmma_f32_16x16x32_f16
// per step (K = H = 32 exactly), one per gate (i,f,g,o), so the gate math is
// per-lane VALU with c kept in 8 VGPRs. h double-buffers through 2KB LDS.
// w_hh fragments (4 x v16h = 32 VGPRs) stay resident for all 1024 steps.
// xp slab for step s+1 (16 rows x 128 gates x f32 = 8KB, contiguous) is
// prefetched with global_load_async_to_lds_b128 (ASYNCcnt) into a
// double-buffered LDS tile, hiding global latency behind WMMA + gate math.
// ---------------------------------------------------------------------------
__global__ __launch_bounds__(64) void lstm_scan(
    const float* __restrict__ xpf, const float* __restrict__ xpr,
    const float* __restrict__ whh_f, const float* __restrict__ whh_r,
    float* __restrict__ out)
{
    const int dir  = blockIdx.x >> 2;      // 0 fwd, 1 bwd
    const int m    = blockIdx.x & 3;       // batch tile
    const float* xp  = dir ? xpr : xpf;
    const float* whh = dir ? whh_r : whh_f;
    const int dirbase = dir * 32;

    const int tid  = threadIdx.x;
    const int lane = tid & 31;
    const int half = tid >> 5;             // wave id = hidden half
    const int lhi  = lane >> 4;            // lane half selects upper K/M block
    const int lmod = lane & 15;

    __shared__ _Float16 hbuf[2][16][32];     // double-buffered h (A-layout src)
    __shared__ float    xbuf[2][16][128];    // double-buffered xp slab (8KB each)

    // B fragments: B[k][n] = w_hh[n][k]; this code's contiguous-K B 32x16 order (same as A):
    // lane (l) element e -> K = (l>=16)*16 + e, N = l&15. 16 contiguous f32.
    v16h bfrag[4];
    #pragma unroll
    for (int g = 0; g < 4; ++g) {
        const int col = (g * 2 + half) * 16 + lmod;   // gate output index
        const float* src = whh + col * 32 + lhi * 16;
        v16h bf;
        #pragma unroll
        for (int e = 0; e < 16; ++e) bf[e] = (_Float16)src[e];
        bfrag[g] = bf;
    }

    // h0 = 0
    for (int i = tid; i < 2 * 16 * 32; i += 64)
        ((_Float16*)hbuf)[i] = (_Float16)0.0f;

    // Async-prefetch xp slab for step 0 into xbuf[0]:
    // 8KB = 64 threads x 8 chunks x 16B.
    {
        const int t0 = dir ? 1023 : 0;
        const unsigned long long g0 =
            (unsigned long long)(xp + ((size_t)t0 * 64 + m * 16) * 128) + tid * 16;
        unsigned lds0 = (unsigned)(unsigned long long)(void*)&xbuf[0][0][0] + tid * 16;
        #pragma unroll
        for (int i = 0; i < 8; ++i) {
            const unsigned long long ga = g0 + (unsigned long long)i * 1024;
            const unsigned la = lds0 + i * 1024;
            asm volatile("global_load_async_to_lds_b128 %0, %1, off"
                         :: "v"(la), "v"(ga) : "memory");
        }
        asm volatile("s_wait_asynccnt 0x0" ::: "memory");
    }
    __syncthreads();

    float cs[8];                                     // cell state tile
    #pragma unroll
    for (int v = 0; v < 8; ++v) cs[v] = 0.0f;

    int cur = 0;
    const int kb = lhi * 16;                         // A K-base for this lane half

    for (int s = 0; s < 1024; ++s) {
        const int t   = dir ? (1023 - s) : s;
        const int nxt = cur ^ 1;

        // Issue async prefetch of next step's xp slab into xbuf[nxt].
        {
            const int sn = (s + 1 < 1024) ? (s + 1) : s;
            const int tn = dir ? (1023 - sn) : sn;
            const unsigned long long g0 =
                (unsigned long long)(xp + ((size_t)tn * 64 + m * 16) * 128) + tid * 16;
            unsigned lds0 = (unsigned)(unsigned long long)(void*)&xbuf[nxt][0][0] + tid * 16;
            #pragma unroll
            for (int i = 0; i < 8; ++i) {
                const unsigned long long ga = g0 + (unsigned long long)i * 1024;
                const unsigned la = lds0 + i * 1024;
                asm volatile("global_load_async_to_lds_b128 %0, %1, off"
                             :: "v"(la), "v"(ga) : "memory");
            }
        }

        // A fragment (16-bit A 16x32): lane<16 -> elems 0..15 = K0..15 ;
        // lane>=16 -> K16..31 (matches B).
        v8h lo = *(const v8h*)&hbuf[cur][lmod][kb];
        v8h hi = *(const v8h*)&hbuf[cur][lmod][kb + 8];
        v16h a;
        #pragma unroll
        for (int e = 0; e < 8; ++e) { a[e] = lo[e]; a[e + 8] = hi[e]; }

        v8f acc[4];
        #pragma unroll
        for (int g = 0; g < 4; ++g) {
            const int n = (g * 2 + half) * 16 + lmod;   // gate column
            v8f cg;                                     // C = xp tile (f32 D layout)
            #pragma unroll
            for (int v = 0; v < 8; ++v)
                cg[v] = xbuf[cur][v + lhi * 8][n];
            acc[g] = __builtin_amdgcn_wmma_f32_16x16x32_f16(
                false, a, false, bfrag[g], (short)0, cg, false, false);
        }

        const int hc = half * 16 + lmod;
        #pragma unroll
        for (int v = 0; v < 8; ++v) {
            const int b = m * 16 + v + lhi * 8;
            const float ig = sigf(acc[0][v]);
            const float fg = sigf(acc[1][v]);
            const float gg = tanh_fast(acc[2][v]);
            const float og = sigf(acc[3][v]);
            const float c  = fg * cs[v] + ig * gg;
            cs[v] = c;
            const float h = og * tanh_fast(c);
            hbuf[nxt][v + lhi * 8][hc] = (_Float16)h;
            out[(b * 1024 + t) * 64 + dirbase + hc] = h;   // [B][T][64]
        }
        // Next-step slab must be resident before anyone reads xbuf[nxt].
        asm volatile("s_wait_asynccnt 0x0" ::: "memory");
        __syncthreads();
        cur = nxt;
    }
}

// ---------------------------------------------------------------------------
// Kernel 3: layer-1 input projections via v_wmma_f32_16x16x4_f32.
// rows r = t*64+b (65536), N=128, K=64. A[r][k] = h0cat[b][t][k].
// ---------------------------------------------------------------------------
__global__ __launch_bounds__(256) void xp1_kernel(
    const float* __restrict__ h0,
    const float* __restrict__ wf, const float* __restrict__ bif, const float* __restrict__ bhf,
    const float* __restrict__ wr, const float* __restrict__ bir, const float* __restrict__ bhr,
    float* __restrict__ xpf, float* __restrict__ xpr)
{
    const int tid = threadIdx.x, lane = tid & 31, wave = tid >> 5;
    const int lhi = lane >> 4, lmod = lane & 15;
    const int unit  = blockIdx.x * 8 + wave;     // 65536 units
    const int ntile = unit & 7;
    const int mtile = (unit >> 3) & 4095;
    const int dir   = unit >> 15;

    const float* w  = dir ? wr : wf;
    const int n     = ntile * 16 + lmod;
    const float bias = dir ? (bir[n] + bhr[n]) : (bif[n] + bhf[n]);

    const int r = mtile * 16 + lmod;             // A row for this lane
    const float* arow = h0 + ((size_t)(r & 63) * 1024 + (r >> 6)) * 64;
    const float* brow = w + (size_t)n * 64;

    v8f acc;
    #pragma unroll
    for (int v = 0; v < 8; ++v) acc[v] = bias;

    #pragma unroll
    for (int k = 0; k < 64; k += 4) {
        // f32 A 16x4: lane elem e -> K = (l>=16)*2 + e ; B symmetric.
        v2f a  = { arow[k + lhi * 2], arow[k + lhi * 2 + 1] };
        v2f bb = { brow[k + lhi * 2], brow[k + lhi * 2 + 1] };
        acc = __builtin_amdgcn_wmma_f32_16x16x4_f32(
            false, a, false, bb, (short)0, acc, false, false);
    }
    float* xpo = dir ? xpr : xpf;
    #pragma unroll
    for (int v = 0; v < 8; ++v)
        xpo[(size_t)(mtile * 16 + v + lhi * 8) * 128 + n] = acc[v];
}

// ---------------------------------------------------------------------------
// Kernel 4: K-split f32 WMMA GEMM partials (used for fc1 and fc2).
// M=64, B row-major [N][K]. Each wave: one (mtile, ntile, kslice) unit,
// accumulates 16x16 f32 tile via v_wmma_f32_16x16x4_f32, writes partial.
// Bandwidth-bound on B (fc1_w = 512 MB f32 -> ~22 us HBM floor).
// ---------------------------------------------------------------------------
__global__ __launch_bounds__(256) void gemm_f32_partial(
    const float* __restrict__ A, const float* __restrict__ B,
    float* __restrict__ P, int N, int K, int kchunk,
    int ngroups, int nkslices)
{
    const int tid = threadIdx.x, lane = tid & 31, wave = tid >> 5;
    const int lhi = lane >> 4, lmod = lane & 15;
    const int blk    = blockIdx.x;
    const int ngroup = blk % ngroups;
    const int kslice = (blk / ngroups) % nkslices;
    const int mtile  = blk / (ngroups * nkslices);
    const int ntile  = ngroup * 8 + wave;
    const int n      = ntile * 16 + lmod;

    const float* arow = A + (size_t)(mtile * 16 + lmod) * K;
    const float* brow = B + (size_t)n * K;
    const int k0 = kslice * kchunk;

    v8f acc = {};
    #pragma unroll 4
    for (int k = k0; k < k0 + kchunk; k += 4) {
        if ((k & 127) == 0) {
            __builtin_prefetch(brow + k + 256, 0, 1);  // global_prefetch_b8
        }
        v2f a  = { arow[k + lhi * 2], arow[k + lhi * 2 + 1] };
        v2f bb = { brow[k + lhi * 2], brow[k + lhi * 2 + 1] };
        acc = __builtin_amdgcn_wmma_f32_16x16x4_f32(
            false, a, false, bb, (short)0, acc, false, false);
    }
    float* prow = P + (size_t)kslice * 64 * N;
    #pragma unroll
    for (int v = 0; v < 8; ++v)
        prow[(size_t)(mtile * 16 + v + lhi * 8) * N + n] = acc[v];
}

// ---------------------------------------------------------------------------
// Kernel 5/6: deterministic partial reductions (+bias, optional relu).
// ---------------------------------------------------------------------------
__global__ __launch_bounds__(256) void reduce_relu(
    const float* __restrict__ P, const float* __restrict__ bias,
    float* __restrict__ out, int N, int slices)
{
    const int idx = blockIdx.x * 256 + threadIdx.x;   // 64*N
    const int n = idx % N;
    float s = bias[n];
    for (int i = 0; i < slices; ++i) s += P[(size_t)i * 64 * N + idx];
    out[idx] = fmaxf(s, 0.0f);
}

__global__ __launch_bounds__(256) void reduce_bias(
    const float* __restrict__ P, const float* __restrict__ bias,
    float* __restrict__ out, int N, int slices)
{
    const int idx = blockIdx.x * 256 + threadIdx.x;
    const int n = idx % N;
    float s = bias[n];
    for (int i = 0; i < slices; ++i) s += P[(size_t)i * 64 * N + idx];
    out[idx] = s;
}

// ---------------------------------------------------------------------------
extern "C" void kernel_launch(void* const* d_in, const int* in_sizes, int n_in,
                              void* d_out, int out_size, void* d_ws, size_t ws_size,
                              hipStream_t stream) {
    const float* x        = (const float*)d_in[0];
    const float* w_ih_l0  = (const float*)d_in[1];
    const float* w_hh_l0  = (const float*)d_in[2];
    const float* b_ih_l0  = (const float*)d_in[3];
    const float* b_hh_l0  = (const float*)d_in[4];
    const float* w_ih_l0r = (const float*)d_in[5];
    const float* w_hh_l0r = (const float*)d_in[6];
    const float* b_ih_l0r = (const float*)d_in[7];
    const float* b_hh_l0r = (const float*)d_in[8];
    const float* w_ih_l1  = (const float*)d_in[9];
    const float* w_hh_l1  = (const float*)d_in[10];
    const float* b_ih_l1  = (const float*)d_in[11];
    const float* b_hh_l1  = (const float*)d_in[12];
    const float* w_ih_l1r = (const float*)d_in[13];
    const float* w_hh_l1r = (const float*)d_in[14];
    const float* b_ih_l1r = (const float*)d_in[15];
    const float* b_hh_l1r = (const float*)d_in[16];
    const float* fc1_w    = (const float*)d_in[17];
    const float* fc1_b    = (const float*)d_in[18];
    const float* fc2_w    = (const float*)d_in[19];
    const float* fc2_b    = (const float*)d_in[20];

    char* ws = (char*)d_ws;
    float* xpf = (float*)(ws + (size_t)0);          // 33.6 MB  [65536][128]
    float* xpr = (float*)(ws + (size_t)33554432);   // 33.6 MB
    float* h0  = (float*)(ws + (size_t)67108864);   // 16.8 MB  [64][1024][64]
    float* h1  = (float*)(ws + (size_t)83886080);   // 16.8 MB
    float* p1  = (float*)(ws + (size_t)100663296);  //  8.4 MB  [16][64][2048]
    float* hid = (float*)(ws + (size_t)109051904);  //  0.5 MB  [64][2048]
    float* p2  = (float*)(ws + (size_t)109576192);  //  9.4 MB  [4][64][9216]

    // layer 0
    xp0_kernel<<<65536, 256, 0, stream>>>(x, w_ih_l0, b_ih_l0, b_hh_l0,
                                          w_ih_l0r, b_ih_l0r, b_hh_l0r, xpf, xpr);
    lstm_scan<<<8, 64, 0, stream>>>(xpf, xpr, w_hh_l0, w_hh_l0r, h0);
    // layer 1 (xp buffers reused; stream order guarantees layer-0 scan done)
    xp1_kernel<<<8192, 256, 0, stream>>>(h0, w_ih_l1, b_ih_l1, b_hh_l1,
                                         w_ih_l1r, b_ih_l1r, b_hh_l1r, xpf, xpr);
    lstm_scan<<<8, 64, 0, stream>>>(xpf, xpr, w_hh_l1, w_hh_l1r, h1);
    // fc1: N=2048 K=65536, 16-way K split -> 4*16*16 = 1024 blocks
    gemm_f32_partial<<<1024, 256, 0, stream>>>(h1, fc1_w, p1, 2048, 65536, 4096, 16, 16);
    reduce_relu<<<512, 256, 0, stream>>>(p1, fc1_b, hid, 2048, 16);
    // fc2: N=9216 K=2048, 4-way K split -> 4*4*72 = 1152 blocks
    gemm_f32_partial<<<1152, 256, 0, stream>>>(hid, fc2_w, p2, 9216, 2048, 512, 72, 4);
    reduce_bias<<<2304, 256, 0, stream>>>(p2, fc2_b, (float*)d_out, 9216, 4);
}
